// NonLocalBlock_36283883717027
// MI455X (gfx1250) — matrix operand order repaired and checked by reading, not compiled, not run on hardware
//
#include <hip/hip_runtime.h>
#include <hip/hip_bf16.h>
#include <math.h>

typedef __attribute__((ext_vector_type(16))) __bf16 v16bf;
typedef __attribute__((ext_vector_type(8)))  __bf16 v8bf;
typedef __attribute__((ext_vector_type(8)))  float  v8f;

union Frag16 { v16bf v; v8bf h[2]; };

__device__ __forceinline__ __bf16 to_bf16(float f) {
  union { float f; unsigned u; } a; a.f = f;
  unsigned r = a.u + 0x7FFFu + ((a.u >> 16) & 1u);   // round-to-nearest-even
  union { unsigned short s; __bf16 b; } o; o.s = (unsigned short)(r >> 16);
  return o.b;
}

#define WMMA_BF16(A, B, C) \
  __builtin_amdgcn_wmma_f32_16x16x32_bf16(false, (A), false, (B), (short)0, (C), false, false)

// ---------------------------------------------------------------- pack weights
__global__ void __launch_bounds__(256)
pack_weights(const float* __restrict__ Wt, const float* __restrict__ Wp,
             const float* __restrict__ Wg, const float* __restrict__ Wo,
             __bf16* __restrict__ Wb3, __bf16* __restrict__ Wob) {
  int idx = blockIdx.x * 256 + threadIdx.x;   // 0..131071
  if (idx < 98304) {                          // 3 x [128][256]
    int which = idx >> 15; int r = idx & 32767;
    const float* src = (which == 0) ? Wt : (which == 1) ? Wp : Wg;
    Wb3[idx] = to_bf16(src[r]);
  } else {                                    // [256][128]
    int r = idx - 98304;
    Wob[r] = to_bf16(Wo[r]);
  }
}

__global__ void zero_buf(float* p, int n) {
  int i = blockIdx.x * 256 + threadIdx.x;
  if (i < n) p[i] = 0.f;
}

// --------------------------------------------------------- QKV projections
// grid (N/64, B, 3), block 128 (4 waves).
// Q,K out: [p][b][n][128].  V out: block-transposed [b][n/64][128][64].
__global__ void __launch_bounds__(128)
proj_kernel(const float* __restrict__ x, const __bf16* __restrict__ Wb3,
            const float* __restrict__ b_theta, const float* __restrict__ b_phi,
            const float* __restrict__ b_g, __bf16* __restrict__ qkv) {
  const int t = threadIdx.x, lane = t & 31, wv = t >> 5;
  const int n0 = blockIdx.x * 64;
  const int b  = blockIdx.y;
  const int p  = blockIdx.z;
  const __bf16* W = Wb3 + (size_t)p * 128 * 256;
  const float* bias = (p == 0) ? b_theta : (p == 1) ? b_phi : b_g;

  __shared__ __bf16 xt[64][40];               // [n][k] for one 32-wide K chunk
  const int row = (wv << 4) + (lane & 15);
  const int b0  = (lane < 16) ? 0 : 8;
  const int hi  = (lane < 16) ? 0 : 16;
  v8f acc[8] = {};

  for (int k0 = 0; k0 < 256; k0 += 32) {
    __syncthreads();
    #pragma unroll
    for (int i = 0; i < 16; ++i) {            // stage x^T tile (transpose via LDS)
      int idx = t + i * 128;
      int nn = idx & 63, kk = idx >> 6;
      xt[nn][kk] = to_bf16(x[((size_t)b * 256 + (k0 + kk)) * 4096 + n0 + nn]);
    }
    __syncthreads();
    Frag16 a;                                 // A: M=n, K=channel
    a.h[0] = *(const v8bf*)&xt[row][hi];
    a.h[1] = *(const v8bf*)&xt[row][hi + 8];
    #pragma unroll
    for (int j = 0; j < 8; ++j) {             // 8 ci-tiles of 16
      const int ci = (j << 4) + (lane & 15);
      const __bf16* wp = W + (size_t)ci * 256 + k0 + hi;
      Frag16 bf;                              // B: N=ci, K contiguous
      bf.h[0] = *(const v8bf*)wp;
      bf.h[1] = *(const v8bf*)(wp + 8);
      acc[j] = WMMA_BF16(a.v, bf.v, acc[j]);
    }
  }
  if (p < 2) {                                // Q/K: [b][n][128]
    __bf16* out = qkv + ((size_t)p * 4 + b) * 4096 * 128;
    const int rbase = n0 + (wv << 4) + ((lane >> 4) << 3);
    #pragma unroll
    for (int j = 0; j < 8; ++j) {
      const int ci = (j << 4) + (lane & 15);
      const float bv = bias[ci];
      #pragma unroll
      for (int v = 0; v < 8; ++v)
        out[(size_t)(rbase + v) * 128 + ci] = to_bf16(acc[j][v] + bv);
    }
  } else {                                    // V: [b][blk][ci][64] (block-transposed)
    __bf16* vout = qkv + ((size_t)8 + b) * 4096 * 128 + (size_t)blockIdx.x * 8192;
    const int mloc = (wv << 4) + ((lane >> 4) << 3);
    #pragma unroll
    for (int j = 0; j < 8; ++j) {
      const int ci = (j << 4) + (lane & 15);
      const float bv = bias[ci];
      #pragma unroll
      for (int v = 0; v < 8; ++v)
        vout[(size_t)ci * 64 + mloc + v] = to_bf16(acc[j][v] + bv);
    }
  }
}

// ------------------------------------------------------------ flash attention
// grid (N/64, B), block 128. Each wave owns 16 query rows, head dim 128.
// K block double-buffered in LDS via gfx1250 async copies.
__global__ void __launch_bounds__(128)
attn_kernel(const __bf16* __restrict__ Q, const __bf16* __restrict__ K,
            const __bf16* __restrict__ Vt, __bf16* __restrict__ O) {
  const int t = threadIdx.x, lane = t & 31, wv = t >> 5;
  const int b = blockIdx.y, n0 = blockIdx.x * 64;
  const int b0 = (lane < 16) ? 0 : 8;
  const int hi = (lane < 16) ? 0 : 16;

  __shared__ __bf16 qs[64][136];      // Q tile  [n][ci]
  __shared__ __bf16 ks[2][64][136];   // K tiles [m][ci], double buffered
  __shared__ __bf16 ps[4][16][72];    // per-wave P tile [n][m]

  const __bf16* Qb  = Q  + ((size_t)b * 4096 + n0) * 128;
  const __bf16* Kb  = K  + (size_t)b * 4096 * 128;
  const __bf16* Vtb = Vt + (size_t)b * 4096 * 128;   // [blk][128][64]

  // stage Q once (vectorized 16B)
  #pragma unroll
  for (int i = 0; i < 8; ++i) {
    int idx = t + i * 128;
    int nn = idx >> 4, cc = (idx & 15) << 3;
    *(v8bf*)&qs[nn][cc] = *(const v8bf*)&Qb[nn * 128 + cc];
  }

  // issue async copy of K block 0 into ks[0] (gfx1250 async-to-LDS path)
  {
    const __bf16* ksrc = Kb;
    const unsigned base = (unsigned)(size_t)&ks[0][0][0];
    #pragma unroll
    for (int cchunk = 0; cchunk < 8; ++cchunk) {
      int cid = t + cchunk * 128;                 // 1024 x 16B chunks
      int r = cid >> 4, c16 = cid & 15;
      unsigned ldsb = base + (unsigned)(r * 272 + c16 * 16);
      unsigned gofs = (unsigned)(r * 256 + c16 * 16);
      asm volatile("global_load_async_to_lds_b128 %0, %1, %2"
                   :: "v"(ldsb), "v"(gofs), "s"(ksrc) : "memory");
    }
  }
  __syncthreads();

  const int row = (wv << 4) + (lane & 15);
  Frag16 qf[4];                    // Q fragments held in registers
  #pragma unroll
  for (int kc = 0; kc < 4; ++kc) {
    qf[kc].h[0] = *(const v8bf*)&qs[row][kc * 32 + hi];
    qf[kc].h[1] = *(const v8bf*)&qs[row][kc * 32 + hi + 8];
  }

  float m_i[8], l_i[8];
  v8f Oacc[8] = {};
  #pragma unroll
  for (int v = 0; v < 8; ++v) { m_i[v] = -1e30f; l_i[v] = 0.f; }

  for (int it = 0; it < 64; ++it) {
    const int par = it & 1;
    asm volatile("s_wait_asynccnt 0x0" ::: "memory");
    __syncthreads();

    // issue async copy of next K block into the other buffer (overlap compute)
    if (it + 1 < 64) {
      const __bf16* ksrc = Kb + (size_t)(it + 1) * 64 * 128;
      const unsigned base = (unsigned)(size_t)&ks[par ^ 1][0][0];
      #pragma unroll
      for (int cchunk = 0; cchunk < 8; ++cchunk) {
        int cid = t + cchunk * 128;
        int r = cid >> 4, c16 = cid & 15;
        unsigned ldsb = base + (unsigned)(r * 272 + c16 * 16);
        unsigned gofs = (unsigned)(r * 256 + c16 * 16);
        asm volatile("global_load_async_to_lds_b128 %0, %1, %2"
                     :: "v"(ldsb), "v"(gofs), "s"(ksrc) : "memory");
      }
    }
    const __bf16* Vblk = Vtb + (size_t)it * 8192;   // [128][64]
    if (it + 1 < 64)                                // prefetch next V block
      __builtin_prefetch(&Vblk[8192 + t * 64], 0, 1);

    // S = Q * K^T  (16x64 per wave), K fragments from LDS
    v8f sacc[4] = {};
    #pragma unroll
    for (int j = 0; j < 4; ++j) {
      const int mloc = (j << 4) + (lane & 15);
      #pragma unroll
      for (int kc = 0; kc < 4; ++kc) {
        Frag16 bf;
        bf.h[0] = *(const v8bf*)&ks[par][mloc][kc * 32 + hi];
        bf.h[1] = *(const v8bf*)&ks[par][mloc][kc * 32 + hi + 8];
        sacc[j] = WMMA_BF16(qf[kc].v, bf.v, sacc[j]);
      }
    }

    // online softmax (rows live in accumulator elements, cols across lanes)
    #pragma unroll
    for (int v = 0; v < 8; ++v) {
      float mx = fmaxf(fmaxf(sacc[0][v], sacc[1][v]), fmaxf(sacc[2][v], sacc[3][v]));
      #pragma unroll
      for (int off = 1; off < 16; off <<= 1) mx = fmaxf(mx, __shfl_xor(mx, off, 32));
      const float mnew = fmaxf(m_i[v], mx);
      const float corr = __expf(m_i[v] - mnew);
      float psum = 0.f;
      #pragma unroll
      for (int j = 0; j < 4; ++j) {
        float pv = __expf(sacc[j][v] - mnew);
        sacc[j][v] = pv;
        psum += pv;
      }
      #pragma unroll
      for (int off = 1; off < 16; off <<= 1) psum += __shfl_xor(psum, off, 32);
      l_i[v] = l_i[v] * corr + psum;
      m_i[v] = mnew;
      #pragma unroll
      for (int jo = 0; jo < 8; ++jo) Oacc[jo][v] *= corr;
    }

    // P: accumulator (D) layout -> A layout via per-wave LDS round-trip
    const int rofs = (lane < 16) ? 0 : 8;
    #pragma unroll
    for (int j = 0; j < 4; ++j)
      #pragma unroll
      for (int v = 0; v < 8; ++v)
        ps[wv][v + rofs][(j << 4) + (lane & 15)] = to_bf16(sacc[j][v]);

    // O += P * V ; V fragments straight from block-transposed global (L2 hits)
    #pragma unroll
    for (int ch = 0; ch < 2; ++ch) {
      Frag16 pa;
      pa.h[0] = *(const v8bf*)&ps[wv][lane & 15][ch * 32 + hi];
      pa.h[1] = *(const v8bf*)&ps[wv][lane & 15][ch * 32 + hi + 8];
      #pragma unroll
      for (int jo = 0; jo < 8; ++jo) {
        const int ci = (jo << 4) + (lane & 15);
        const __bf16* vp = Vblk + (size_t)ci * 64 + ch * 32 + hi;
        Frag16 vf;
        vf.h[0] = *(const v8bf*)vp;
        vf.h[1] = *(const v8bf*)(vp + 8);
        Oacc[jo] = WMMA_BF16(pa.v, vf.v, Oacc[jo]);
      }
    }
  }

  __bf16* Ob = O + ((size_t)b * 4096 + n0) * 128;
  const int rbase = (wv << 4) + ((lane >> 4) << 3);
  #pragma unroll
  for (int jo = 0; jo < 8; ++jo) {
    const int ci = (jo << 4) + (lane & 15);
    #pragma unroll
    for (int v = 0; v < 8; ++v)
      Ob[(size_t)(rbase + v) * 128 + ci] = to_bf16(Oacc[jo][v] / l_i[v]);
  }
}

// ------------------------------------------- output projection + BN statistics
// grid (N/64, B), block 128. y[b][c][n] fp32 + per-channel sum/sumsq atomics.
__global__ void __launch_bounds__(128)
outproj_kernel(const __bf16* __restrict__ O, const __bf16* __restrict__ Wo,
               const float* __restrict__ b_out, float* __restrict__ Y,
               float* __restrict__ sums) {
  const int t = threadIdx.x, lane = t & 31, wv = t >> 5;
  const int n0 = blockIdx.x * 64, b = blockIdx.y;
  const int b0 = (lane < 16) ? 0 : 8;
  const int hi = (lane < 16) ? 0 : 16;
  const int nrow = n0 + (wv << 4) + (lane & 15);
  const __bf16* Orow = O + ((size_t)b * 4096 + nrow) * 128;

  v8f acc[16] = {};
  #pragma unroll
  for (int kc = 0; kc < 4; ++kc) {
    Frag16 a;                                  // A direct from global (ci-contig)
    a.h[0] = *(const v8bf*)&Orow[kc * 32 + hi];
    a.h[1] = *(const v8bf*)&Orow[kc * 32 + hi + 8];
    #pragma unroll
    for (int j = 0; j < 16; ++j) {
      const int c = (j << 4) + (lane & 15);
      const __bf16* wp = Wo + (size_t)c * 128 + kc * 32 + hi;
      Frag16 bf;
      bf.h[0] = *(const v8bf*)wp;
      bf.h[1] = *(const v8bf*)(wp + 8);
      acc[j] = WMMA_BF16(a.v, bf.v, acc[j]);
    }
  }
  const int rbase = n0 + (wv << 4) + ((lane >> 4) << 3);
  #pragma unroll
  for (int j = 0; j < 16; ++j) {
    const int c = (j << 4) + (lane & 15);
    const float bv = b_out[c];
    float s1 = 0.f, s2 = 0.f;
    #pragma unroll
    for (int v = 0; v < 8; ++v) {
      float yv = acc[j][v] + bv;
      Y[((size_t)b * 256 + c) * 4096 + rbase + v] = yv;
      s1 += yv; s2 += yv * yv;
    }
    s1 += __shfl_xor(s1, 16, 32);              // partner lane shares channel c
    s2 += __shfl_xor(s2, 16, 32);
    if (lane < 16) {
      atomicAdd(&sums[c], s1);
      atomicAdd(&sums[256 + c], s2);
    }
  }
}

// --------------------------------------------------- BN normalize + residual
__global__ void __launch_bounds__(256)
finalize_kernel(const float* __restrict__ x, const float* __restrict__ Y,
                const float* __restrict__ sums, const float* __restrict__ gamma,
                const float* __restrict__ beta, const float* __restrict__ scale,
                float* __restrict__ out) {
  int idx = blockIdx.x * 256 + threadIdx.x;    // [b][c][n] linear, 4*256*4096
  int c = (idx >> 12) & 255;
  float mean = sums[c] * (1.0f / 16384.0f);
  float var  = sums[256 + c] * (1.0f / 16384.0f) - mean * mean;
  float r = rsqrtf(var + 1e-5f);
  float y = (Y[idx] - mean) * r * gamma[c] + beta[c];
  out[idx] = x[idx] + y * scale[0];
}

// ------------------------------------------------------------------ launcher
extern "C" void kernel_launch(void* const* d_in, const int* in_sizes, int n_in,
                              void* d_out, int out_size, void* d_ws, size_t ws_size,
                              hipStream_t stream) {
  (void)in_sizes; (void)n_in; (void)out_size; (void)ws_size;
  const float* x       = (const float*)d_in[0];
  const float* W_theta = (const float*)d_in[1];
  const float* b_theta = (const float*)d_in[2];
  const float* W_phi   = (const float*)d_in[3];
  const float* b_phi   = (const float*)d_in[4];
  const float* W_g     = (const float*)d_in[5];
  const float* b_g     = (const float*)d_in[6];
  const float* W_out   = (const float*)d_in[7];
  const float* b_out   = (const float*)d_in[8];
  const float* gamma   = (const float*)d_in[9];
  const float* beta    = (const float*)d_in[10];
  const float* scale   = (const float*)d_in[11];
  float* out = (float*)d_out;

  char* ws = (char*)d_ws;
  __bf16* Wb3  = (__bf16*)(ws);                 // 3*128*256 bf16
  __bf16* Wob  = (__bf16*)(ws + 196608);        // 256*128 bf16
  __bf16* QKV  = (__bf16*)(ws + 262144);        // 3*B*N*128 bf16
  __bf16* Obf  = (__bf16*)(ws + 12845056);      // B*N*128 bf16
  float*  Y    = (float*) (ws + 17039360);      // B*256*N f32
  float*  sums = (float*) (ws + 33816576);      // 2*256 f32
  __bf16* Qp = QKV;
  __bf16* Kp = QKV + (size_t)4 * 4096 * 128;
  __bf16* Vp = QKV + (size_t)8 * 4096 * 128;    // block-transposed layout

  pack_weights<<<512, 256, 0, stream>>>(W_theta, W_phi, W_g, W_out, Wb3, Wob);
  zero_buf<<<2, 256, 0, stream>>>(sums, 512);
  proj_kernel<<<dim3(64, 4, 3), 128, 0, stream>>>(x, Wb3, b_theta, b_phi, b_g, QKV);
  attn_kernel<<<dim3(64, 4), 128, 0, stream>>>(Qp, Kp, Vp, Obf);
  outproj_kernel<<<dim3(64, 4), 128, 0, stream>>>(Obf, Wob, b_out, Y, sums);
  finalize_kernel<<<16384, 256, 0, stream>>>(x, Y, sums, gamma, beta, scale, out);
}
